// NonLocalBlock_14645838479373
// MI455X (gfx1250) — matrix operand order repaired and checked by reading, not compiled, not run on hardware
//
#include <hip/hip_runtime.h>
#include <cstdint>

#define DEVI __device__ __forceinline__

typedef __attribute__((ext_vector_type(16))) __bf16 v16bf;
typedef __attribute__((ext_vector_type(8)))  float  v8f;

static constexpr int C = 256;
static constexpr int N = 8192;          // 8*32*32
static constexpr float EPS = 1e-5f;

union Frag {
    v16bf v;
    uint4 q[2];
    unsigned short u[16];
};

DEVI unsigned short f2bf(float f) {
    union { float f; uint32_t u; } cvt; cvt.f = f;
    uint32_t u = cvt.u;
    u += 0x7FFFu + ((u >> 16) & 1u);    // round to nearest even
    return (unsigned short)(u >> 16);
}

DEVI v8f vzero() {
    v8f z;
#pragma unroll
    for (int i = 0; i < 8; i++) z[i] = 0.0f;
    return z;
}

DEVI v8f wmma_bf16(const Frag& a, const Frag& b, v8f c) {
    return __builtin_amdgcn_wmma_f32_16x16x32_bf16(
        /*neg_a=*/false, a.v, /*neg_b=*/false, b.v,
        /*c_mod=*/(short)0, c, /*reuse_a=*/false, /*reuse_b=*/false);
}

// async global->LDS copy of 16 bytes per lane (tracked by ASYNCcnt)
DEVI void async_copy_b128(uint32_t lds_off, uint64_t gaddr) {
    asm volatile("global_load_async_to_lds_b128 %0, %1, off"
                 :: "v"(lds_off), "v"(gaddr) : "memory");
}

// ---------------------------------------------------------------- GroupNorm
__global__ void gn_partial_kernel(const float* __restrict__ x,
                                  float* __restrict__ psum, float* __restrict__ psq) {
    __shared__ float ss[256], s2[256];
    int tid = threadIdx.x;
    int idx = blockIdx.x * 256 + tid;
    const int total = C * N;
    float a = 0.0f, b = 0.0f;
    for (int i = idx; i < total; i += 256 * 256) {
        float v = x[i]; a += v; b += v * v;
    }
    ss[tid] = a; s2[tid] = b;
    __syncthreads();
    for (int s = 128; s > 0; s >>= 1) {
        if (tid < s) { ss[tid] += ss[tid + s]; s2[tid] += s2[tid + s]; }
        __syncthreads();
    }
    if (tid == 0) { psum[blockIdx.x] = ss[0]; psq[blockIdx.x] = s2[0]; }
}

__global__ void gn_final_kernel(const float* __restrict__ psum,
                                const float* __restrict__ psq,
                                float* __restrict__ stats) {
    __shared__ float ss[256], s2[256];
    int tid = threadIdx.x;
    ss[tid] = psum[tid]; s2[tid] = psq[tid];
    __syncthreads();
    for (int s = 128; s > 0; s >>= 1) {
        if (tid < s) { ss[tid] += ss[tid + s]; s2[tid] += s2[tid + s]; }
        __syncthreads();
    }
    if (tid == 0) {
        float inv = 1.0f / ((float)C * (float)N);
        float mean = ss[0] * inv;
        float var  = s2[0] * inv - mean * mean;
        stats[0] = mean;
        stats[1] = rsqrtf(var + EPS);
    }
}

// normalize + affine; write xn f32 [C,N] and xnT bf16 [N,C] via LDS transpose
__global__ void gn_normalize_kernel(const float* __restrict__ x,
                                    const float* __restrict__ gw,
                                    const float* __restrict__ gb,
                                    const float* __restrict__ stats,
                                    float* __restrict__ xn,
                                    unsigned short* __restrict__ xnT) {
    __shared__ unsigned short sh[32][33];
    const float mean = stats[0], rstd = stats[1];
    int tile = blockIdx.x;            // (C/32)*(N/32) tiles
    int tc = tile >> 8;               // 0..7   (N/32 == 256)
    int tn = tile & 255;              // 0..255
    int c0 = tc * 32, n0 = tn * 32;
    int col = threadIdx.x & 31;
    int row = threadIdx.x >> 5;       // 0..7
#pragma unroll
    for (int i = 0; i < 4; i++) {
        int cl = row + 8 * i;
        int c = c0 + cl;
        int idx = c * N + n0 + col;
        float o = (x[idx] - mean) * rstd * gw[c] + gb[c];
        xn[idx] = o;
        sh[cl][col] = f2bf(o);
    }
    __syncthreads();
#pragma unroll
    for (int i = 0; i < 4; i++) {
        int nl = row + 8 * i;
        xnT[(n0 + nl) * C + c0 + col] = sh[col][nl];
    }
}

// -------------------------------------------------- weight f32 -> bf16 copies
__global__ void cvt4_kernel(const float* __restrict__ a, const float* __restrict__ b,
                            const float* __restrict__ c, const float* __restrict__ d,
                            unsigned short* __restrict__ oa, unsigned short* __restrict__ ob_,
                            unsigned short* __restrict__ oc, unsigned short* __restrict__ od) {
    int i = blockIdx.x * 256 + threadIdx.x;   // grid 256 -> 65536 = C*C
    oa[i] = f2bf(a[i]); ob_[i] = f2bf(b[i]);
    oc[i] = f2bf(c[i]); od[i]  = f2bf(d[i]);
}

// ---------------------------------------------------------------- QKV GEMMs
// qT/kT stored [N,C] bf16 (scores B operands contiguous); v stored [C,N] bf16.
// 1/sqrt(C)=1/16 folded into qT.
__global__ void qkv_proj_kernel(const unsigned short* __restrict__ xnT,
                                const unsigned short* __restrict__ wq, const float* __restrict__ qb,
                                const unsigned short* __restrict__ wk, const float* __restrict__ kb,
                                const unsigned short* __restrict__ wv, const float* __restrict__ vb,
                                unsigned short* __restrict__ qT,
                                unsigned short* __restrict__ kT,
                                unsigned short* __restrict__ vC) {
    int lane = threadIdx.x & 31;
    int wid  = threadIdx.x >> 5;
    int tile = blockIdx.x * 8 + wid;       // 8192 wave-tiles
    int n0 = (tile >> 4) * 16;
    int o0 = (tile & 15) * 16;
    int lm = lane & 15;
    bool lo = lane < 16;
    const int ka1 = lo ? 0 : 16, ka2 = lo ? 8 : 24, kb0 = lo ? 0 : 16;

    Frag A[8];
#pragma unroll
    for (int cb = 0; cb < 8; cb++) {
        const unsigned short* ap = xnT + (n0 + lm) * C + cb * 32;
        A[cb].q[0] = *(const uint4*)(ap + ka1);
        A[cb].q[1] = *(const uint4*)(ap + ka2);
    }

    v8f aq = vzero(), ak = vzero(), av = vzero();
#pragma unroll
    for (int cb = 0; cb < 8; cb++) {
        Frag B;
        const unsigned short* bp = wq + (o0 + lm) * C + cb * 32 + kb0;
        B.q[0] = *(const uint4*)bp; B.q[1] = *(const uint4*)(bp + 8);
        aq = wmma_bf16(A[cb], B, aq);
        bp = wk + (o0 + lm) * C + cb * 32 + kb0;
        B.q[0] = *(const uint4*)bp; B.q[1] = *(const uint4*)(bp + 8);
        ak = wmma_bf16(A[cb], B, ak);
        bp = wv + (o0 + lm) * C + cb * 32 + kb0;
        B.q[0] = *(const uint4*)bp; B.q[1] = *(const uint4*)(bp + 8);
        av = wmma_bf16(A[cb], B, av);
    }

    float bq = qb[o0 + lm], bk = kb[o0 + lm], bv = vb[o0 + lm];
    int mb = lo ? 0 : 8;
#pragma unroll
    for (int r = 0; r < 8; r++) {
        int m = r + mb;
        qT[(n0 + m) * C + o0 + lm] = f2bf((aq[r] + bq) * 0.0625f);
        kT[(n0 + m) * C + o0 + lm] = f2bf(ak[r] + bk);
    }
    union { uint4 q; unsigned short u[8]; } pk;
#pragma unroll
    for (int r = 0; r < 8; r++) pk.u[r] = f2bf(av[r] + bv);
    *(uint4*)(vC + (o0 + lm) * N + n0 + mb) = pk.q;
}

// ------------------------------------------------------- flash attention
// Block = 8 waves = 4 query tiles x 2 channel halves; all waves share the same
// key loop, so K/V tiles are staged in LDS via async copies, double-buffered.
__global__ void flash_attn_kernel(const unsigned short* __restrict__ qT,
                                  const unsigned short* __restrict__ kT,
                                  const unsigned short* __restrict__ vC,
                                  unsigned short* __restrict__ attT) {
    __shared__ unsigned short kbuf[2][32 * 256];   // [key][c]  16KB x2
    __shared__ unsigned short vbuf[2][256 * 32];   // [c][key]  16KB x2
    __shared__ unsigned short Pb[8][16][32];       // per-wave P staging, 8KB

    int tid  = threadIdx.x;
    int lane = tid & 31;
    int wid  = tid >> 5;
    int wt   = blockIdx.x * 8 + wid;               // 1024 wave-tiles
    int nt    = wt >> 1;                           // query tile 0..511
    int chalf = wt & 1;                            // channel half 0..1
    int n0 = nt * 16;
    int lm = lane & 15;
    bool lo = lane < 16;
    const int ka1 = lo ? 0 : 16, ka2 = lo ? 8 : 24, kb0 = lo ? 0 : 16;
    int mb = lo ? 0 : 8;

    Frag Qf[8];
#pragma unroll
    for (int cb = 0; cb < 8; cb++) {
        const unsigned short* ap = qT + (n0 + lm) * C + cb * 32;
        Qf[cb].q[0] = *(const uint4*)(ap + ka1);
        Qf[cb].q[1] = *(const uint4*)(ap + ka2);
    }

    v8f acc[8];
#pragma unroll
    for (int t = 0; t < 8; t++) acc[t] = vzero();
    float mrow[8], lrow[8];
#pragma unroll
    for (int r = 0; r < 8; r++) { mrow[r] = -1e30f; lrow[r] = 0.0f; }

    const int NKT = N / 32;                        // 256 key tiles

    // issue async copy of key-tile kt into buffer buf (8 b128 per thread)
    auto issue_tile = [&](int kt, int buf) {
        const unsigned short* ksrc = kT + kt * 32 * C;   // 16KB contiguous
        uint32_t klds = (uint32_t)(uintptr_t)&kbuf[buf][0];
#pragma unroll
        for (int j = 0; j < 4; j++) {
            int chunk = tid + j * 256;                   // 0..1023
            async_copy_b128(klds + chunk * 16,
                            (uint64_t)(uintptr_t)(ksrc + chunk * 8));
        }
        const unsigned short* vsrc = vC + tid * N + kt * 32;  // row c = tid
        uint32_t vlds = (uint32_t)(uintptr_t)&vbuf[buf][tid * 32];
#pragma unroll
        for (int j = 0; j < 4; j++) {
            async_copy_b128(vlds + j * 16,
                            (uint64_t)(uintptr_t)(vsrc + j * 8));
        }
    };

    issue_tile(0, 0);

    for (int it = 0; it < NKT; it++) {
        int cur = it & 1;
        if (it + 1 < NKT) {
            issue_tile(it + 1, cur ^ 1);
            asm volatile("s_wait_asynccnt 0x8" ::: "memory");  // current tile done
        } else {
            asm volatile("s_wait_asynccnt 0x0" ::: "memory");
        }
        __syncthreads();   // all waves' chunks of current tile visible

        // S = q^T k for 16 queries x 32 keys (two 16x16 tiles), K=C=256
        v8f S0 = vzero(), S1 = vzero();
#pragma unroll
        for (int cb = 0; cb < 8; cb++) {
            Frag B0, B1;
            const unsigned short* b0 = &kbuf[cur][lm * C + cb * 32 + kb0];
            const unsigned short* b1 = &kbuf[cur][(16 + lm) * C + cb * 32 + kb0];
            B0.q[0] = *(const uint4*)b0; B0.q[1] = *(const uint4*)(b0 + 8);
            B1.q[0] = *(const uint4*)b1; B1.q[1] = *(const uint4*)(b1 + 8);
            S0 = wmma_bf16(Qf[cb], B0, S0);
            S1 = wmma_bf16(Qf[cb], B1, S1);
        }
        // online softmax; row m = r + mb lives across the 16-lane half
#pragma unroll
        for (int r = 0; r < 8; r++) {
            float t = fmaxf(S0[r], S1[r]);
            t = fmaxf(t, __shfl_xor(t, 1));
            t = fmaxf(t, __shfl_xor(t, 2));
            t = fmaxf(t, __shfl_xor(t, 4));
            t = fmaxf(t, __shfl_xor(t, 8));
            float mnew  = fmaxf(mrow[r], t);
            float alpha = __expf(mrow[r] - mnew);
            float p0 = __expf(S0[r] - mnew);
            float p1 = __expf(S1[r] - mnew);
            float rs = p0 + p1;
            rs += __shfl_xor(rs, 1);
            rs += __shfl_xor(rs, 2);
            rs += __shfl_xor(rs, 4);
            rs += __shfl_xor(rs, 8);
            lrow[r] = lrow[r] * alpha + rs;
            mrow[r] = mnew;
#pragma unroll
            for (int t2 = 0; t2 < 8; t2++) acc[t2][r] *= alpha;
            int m = r + mb;
            Pb[wid][m][lm]      = f2bf(p0);
            Pb[wid][m][16 + lm] = f2bf(p1);
        }
        // P (C-layout) -> A-layout through LDS
        Frag Pa;
        Pa.q[0] = *(const uint4*)&Pb[wid][lm][ka1];
        Pa.q[1] = *(const uint4*)&Pb[wid][lm][ka2];
        // acc += P * V   (K = 32 keys), 8 channel tiles for this wave
#pragma unroll
        for (int t2 = 0; t2 < 8; t2++) {
            int c0 = (chalf * 8 + t2) * 16;
            Frag Bv;
            const unsigned short* bp = &vbuf[cur][(c0 + lm) * 32 + kb0];
            Bv.q[0] = *(const uint4*)bp; Bv.q[1] = *(const uint4*)(bp + 8);
            acc[t2] = wmma_bf16(Pa, Bv, acc[t2]);
        }
        __syncthreads();   // compute done before buffer is overwritten
    }

#pragma unroll
    for (int r = 0; r < 8; r++) lrow[r] = 1.0f / lrow[r];
#pragma unroll
    for (int t2 = 0; t2 < 8; t2++) {
        int c0 = (chalf * 8 + t2) * 16;
#pragma unroll
        for (int r = 0; r < 8; r++) {
            int m = r + mb;
            attT[(n0 + m) * C + c0 + lm] = f2bf(acc[t2][r] * lrow[r]);
        }
    }
}

// ------------------------------------------ output projection + residual
__global__ void out_proj_kernel(const unsigned short* __restrict__ attT,
                                const unsigned short* __restrict__ wo,
                                const float* __restrict__ ob,
                                const float* __restrict__ xn,
                                float* __restrict__ out) {
    int lane = threadIdx.x & 31;
    int wid  = threadIdx.x >> 5;
    int tile = blockIdx.x * 8 + wid;
    int n0 = (tile >> 4) * 16;
    int c0 = (tile & 15) * 16;
    int lm = lane & 15;
    bool lo = lane < 16;
    const int ka1 = lo ? 0 : 16, ka2 = lo ? 8 : 24, kb0 = lo ? 0 : 16;

    v8f acc = vzero();
#pragma unroll
    for (int cb = 0; cb < 8; cb++) {
        Frag A, B;
        const unsigned short* ap = attT + (n0 + lm) * C + cb * 32;
        A.q[0] = *(const uint4*)(ap + ka1);
        A.q[1] = *(const uint4*)(ap + ka2);
        const unsigned short* bp = wo + (c0 + lm) * C + cb * 32 + kb0;
        B.q[0] = *(const uint4*)bp; B.q[1] = *(const uint4*)(bp + 8);
        acc = wmma_bf16(A, B, acc);
    }
    float bias = ob[c0 + lm];
    int base = (c0 + lm) * N + n0 + (lo ? 0 : 8);
    const float4* xr = (const float4*)(xn + base);
    float4 r0 = xr[0], r1 = xr[1];
    float4 o0v, o1v;
    o0v.x = r0.x + acc[0] + bias; o0v.y = r0.y + acc[1] + bias;
    o0v.z = r0.z + acc[2] + bias; o0v.w = r0.w + acc[3] + bias;
    o1v.x = r1.x + acc[4] + bias; o1v.y = r1.y + acc[5] + bias;
    o1v.z = r1.z + acc[6] + bias; o1v.w = r1.w + acc[7] + bias;
    *(float4*)(out + base)     = o0v;
    *(float4*)(out + base + 4) = o1v;
}

// ------------------------------------------------------------------ launch
extern "C" void kernel_launch(void* const* d_in, const int* in_sizes, int n_in,
                              void* d_out, int out_size, void* d_ws, size_t ws_size,
                              hipStream_t stream) {
    const float* x  = (const float*)d_in[0];
    const float* gw = (const float*)d_in[1];
    const float* gb = (const float*)d_in[2];
    const float* qw = (const float*)d_in[3];
    const float* qb = (const float*)d_in[4];
    const float* kw = (const float*)d_in[5];
    const float* kb = (const float*)d_in[6];
    const float* vw = (const float*)d_in[7];
    const float* vb = (const float*)d_in[8];
    const float* ow = (const float*)d_in[9];
    const float* ob = (const float*)d_in[10];
    float* out = (float*)d_out;

    float* wsf   = (float*)d_ws;
    float* psum  = wsf;                 // 256 f32
    float* psq   = wsf + 256;           // 256 f32
    float* stats = wsf + 512;           // mean, rstd
    float* xn    = wsf + 1024;          // C*N f32 (8 MB)

    unsigned short* bf  = (unsigned short*)(wsf + 1024 + (size_t)C * N);
    unsigned short* xnT  = bf;                         // [N,C] bf16
    unsigned short* qT   = xnT  + (size_t)N * C;       // [N,C]
    unsigned short* kT   = qT   + (size_t)N * C;       // [N,C]
    unsigned short* vC   = kT   + (size_t)N * C;       // [C,N]
    unsigned short* attT = vC   + (size_t)N * C;       // [N,C]
    unsigned short* wqb  = attT + (size_t)N * C;       // C*C each
    unsigned short* wkb  = wqb + C * C;
    unsigned short* wvb  = wkb + C * C;
    unsigned short* wob  = wvb + C * C;

    gn_partial_kernel<<<256, 256, 0, stream>>>(x, psum, psq);
    gn_final_kernel<<<1, 256, 0, stream>>>(psum, psq, stats);
    gn_normalize_kernel<<<(C / 32) * (N / 32), 256, 0, stream>>>(x, gw, gb, stats, xn, xnT);
    cvt4_kernel<<<C * C / 256, 256, 0, stream>>>(qw, kw, vw, ow, wqb, wkb, wvb, wob);
    qkv_proj_kernel<<<(N / 16) * (C / 16) / 8, 256, 0, stream>>>(
        xnT, wqb, qb, wkb, kb, wvb, vb, qT, kT, vC);
    flash_attn_kernel<<<(N / 16) * 2 / 8, 256, 0, stream>>>(qT, kT, vC, attT);
    out_proj_kernel<<<(N / 16) * (C / 16) / 8, 256, 0, stream>>>(attT, wob, ob, xn, out);
}
